// RNNStateEncoder_57071525429935
// MI455X (gfx1250) — matrix operand order repaired and checked by reading, not compiled, not run on hardware
//
#include <hip/hip_runtime.h>
#include <hip/hip_bf16.h>

// GRU seq_forward for T=128, N=256, I=H=512 on gfx1250 (MI455X).
// bf16 WMMA (v_wmma_f32_16x16x32_bf16) with f32 accumulation.
// Prep pass converts x / w_ih / w_hh to bf16 in scratch, seeds the f32+bf16
// hidden-state ping-pong buffers, and zeroes a "zero row" used to implement
// the episode-reset mask branch-free. Then 128 stream-ordered step kernels
// each fuse BOTH GEMMs (x_t @ w_ih^T and masked-h @ w_hh^T) + gate math.

typedef __attribute__((ext_vector_type(16))) __bf16 v16bf;
typedef __attribute__((ext_vector_type(8)))  __bf16 v8bf;
typedef __attribute__((ext_vector_type(8)))  float  v8f;

#define T_DIM 128
#define N_DIM 256
#define I_DIM 512
#define H_DIM 512

// ---------------------------------------------------------------- prep kernels

__global__ __launch_bounds__(256) void cvt_f32_to_bf16(const float* __restrict__ src,
                                                       __bf16* __restrict__ dst, int n) {
    for (int i = blockIdx.x * blockDim.x + threadIdx.x; i < n; i += gridDim.x * blockDim.x)
        dst[i] = (__bf16)src[i];
}

__global__ __launch_bounds__(256) void init_hidden(const float* __restrict__ h0,
                                                   float* __restrict__ hf,
                                                   __bf16* __restrict__ hb,
                                                   __bf16* __restrict__ zrow, int n) {
    int i = blockIdx.x * blockDim.x + threadIdx.x;
    if (i < n) {
        float v = h0[i];
        hf[i] = v;
        hb[i] = (__bf16)v;
    }
    if (i < H_DIM) zrow[i] = (__bf16)0.0f;   // zero row for masked episodes
}

// ---------------------------------------------------------------- step kernel

// Grid: 512 waves (16 row-tiles over N x 32 col-tiles over H), 8 waves/block.
// Each wave owns one 16x16 output tile of h_new and carries 6 f32 WMMA
// accumulators: {x,h} x {r,z,n} gates.
__global__ __launch_bounds__(256) void gru_step_kernel(
    const __bf16* __restrict__ xbf,      // (T*N, I) bf16
    const __bf16* __restrict__ wih,      // (3H, I)  bf16
    const __bf16* __restrict__ whh,      // (3H, H)  bf16
    const float*  __restrict__ b_ih,     // (3H,)
    const float*  __restrict__ b_hh,     // (3H,)
    const unsigned char* __restrict__ masks, // (T*N,) jax bool = 1 byte
    const float*  __restrict__ h_in_f,   // (N,H) f32 previous hidden
    const __bf16* __restrict__ h_in_b,   // (N,H) bf16 previous hidden
    const __bf16* __restrict__ zrow,     // (H,) bf16 zeros
    float*  __restrict__ h_out_f,        // (N,H) f32 next hidden
    __bf16* __restrict__ h_out_b,        // (N,H) bf16 next hidden
    float*  __restrict__ outs,           // d_out + t*N*H
    float*  __restrict__ h_final,        // d_out + T*N*H (only for t==T-1)
    int t)
{
    const int lane = threadIdx.x & 31;
    const int wid  = (blockIdx.x * blockDim.x + threadIdx.x) >> 5;
    const int rowBase = (wid & 15) << 4;   // over N
    const int colBase = (wid >> 4) << 4;   // over H

    const int m16  = lane & 15;
    const int half = lane >> 4;

    // ---- A-fragment sources: row of x_t and (masked) row of h_prev.
    // Branch-free episode reset: masked lanes read the shared zero row.
    const int  arow = rowBase + m16;
    const bool am   = masks[t * N_DIM + arow] != 0;
    const __bf16* xrow = xbf + (size_t)(t * N_DIM + arow) * I_DIM;
    const __bf16* hrow = am ? (h_in_b + (size_t)arow * H_DIM) : zrow;

    // ---- B-fragment sources: weight rows for this output column, 3 gates
    const int bcol = colBase + m16;
    const __bf16* wiR = wih + (size_t)(bcol            ) * I_DIM;
    const __bf16* wiZ = wih + (size_t)(bcol +     H_DIM) * I_DIM;
    const __bf16* wiN = wih + (size_t)(bcol + 2 * H_DIM) * I_DIM;
    const __bf16* whR = whh + (size_t)(bcol            ) * H_DIM;
    const __bf16* whZ = whh + (size_t)(bcol +     H_DIM) * H_DIM;
    const __bf16* whN = whh + (size_t)(bcol + 2 * H_DIM) * H_DIM;

    v8f ciR = {}, ciZ = {}, ciN = {};   // x-side gate pre-activations
    v8f chR = {}, chZ = {}, chN = {};   // h-side gate pre-activations

    for (int k0 = 0; k0 < I_DIM; k0 += 32) {
        const int aLo = k0 + 16 * half;       // A slots 0-7
        const int aHi = k0 + 16 * half + 8;   // A slots 8-15
        const int bOf = k0 + 16 * half;       // B slots 0-15 (contiguous)

        v8bf xlo = *(const v8bf*)(xrow + aLo);
        v8bf xhi = *(const v8bf*)(xrow + aHi);
        v16bf ax = __builtin_shufflevector(xlo, xhi,
                       0,1,2,3,4,5,6,7,8,9,10,11,12,13,14,15);

        v8bf hlo = *(const v8bf*)(hrow + aLo);
        v8bf hhi = *(const v8bf*)(hrow + aHi);
        v16bf ah = __builtin_shufflevector(hlo, hhi,
                       0,1,2,3,4,5,6,7,8,9,10,11,12,13,14,15);

        v16bf biR = *(const v16bf*)(wiR + bOf);
        v16bf biZ = *(const v16bf*)(wiZ + bOf);
        v16bf biN = *(const v16bf*)(wiN + bOf);
        v16bf bhR = *(const v16bf*)(whR + bOf);
        v16bf bhZ = *(const v16bf*)(whZ + bOf);
        v16bf bhN = *(const v16bf*)(whN + bOf);

        ciR = __builtin_amdgcn_wmma_f32_16x16x32_bf16(false, ax, false, biR, (short)0, ciR, false, false);
        ciZ = __builtin_amdgcn_wmma_f32_16x16x32_bf16(false, ax, false, biZ, (short)0, ciZ, false, false);
        ciN = __builtin_amdgcn_wmma_f32_16x16x32_bf16(false, ax, false, biN, (short)0, ciN, false, false);
        chR = __builtin_amdgcn_wmma_f32_16x16x32_bf16(false, ah, false, bhR, (short)0, chR, false, false);
        chZ = __builtin_amdgcn_wmma_f32_16x16x32_bf16(false, ah, false, bhZ, (short)0, chZ, false, false);
        chN = __builtin_amdgcn_wmma_f32_16x16x32_bf16(false, ah, false, bhN, (short)0, chN, false, false);
    }

    // ---- gate combine. C/D layout: VGPR j -> row rowBase+j+8*half, col = colBase+(lane&15)
    const int   ncol = colBase + m16;
    const float birC = b_ih[ncol], bizC = b_ih[ncol + H_DIM], binC = b_ih[ncol + 2 * H_DIM];
    const float bhrC = b_hh[ncol], bhzC = b_hh[ncol + H_DIM], bhnC = b_hh[ncol + 2 * H_DIM];

#pragma unroll
    for (int j = 0; j < 8; ++j) {
        const int mrow = rowBase + j + 8 * half;
        const float r  = 1.0f / (1.0f + __expf(-(ciR[j] + birC + chR[j] + bhrC)));
        const float z  = 1.0f / (1.0f + __expf(-(ciZ[j] + bizC + chZ[j] + bhzC)));
        const float nn = tanhf(ciN[j] + binC + r * (chN[j] + bhnC));

        // unconditional load + cndmask select (address always valid)
        float hp = h_in_f[(size_t)mrow * H_DIM + ncol];
        if (masks[t * N_DIM + mrow] == 0) hp = 0.0f;
        const float hn = (1.0f - z) * nn + z * hp;

        outs[(size_t)mrow * H_DIM + ncol]    = hn;
        h_out_f[(size_t)mrow * H_DIM + ncol] = hn;
        h_out_b[(size_t)mrow * H_DIM + ncol] = (__bf16)hn;
        if (h_final) h_final[(size_t)mrow * H_DIM + ncol] = hn;
    }
}

// ---------------------------------------------------------------- launch

extern "C" void kernel_launch(void* const* d_in, const int* in_sizes, int n_in,
                              void* d_out, int out_size, void* d_ws, size_t ws_size,
                              hipStream_t stream) {
    (void)in_sizes; (void)n_in; (void)out_size; (void)ws_size;

    const float* x    = (const float*)d_in[0];              // (T*N, I)
    const float* h0   = (const float*)d_in[1];              // (1, N, H)
    const unsigned char* masks = (const unsigned char*)d_in[2]; // (T*N,) bool
    const float* w_ih = (const float*)d_in[3];              // (3H, I)
    const float* w_hh = (const float*)d_in[4];              // (3H, H)
    const float* b_ih = (const float*)d_in[5];              // (3H,)
    const float* b_hh = (const float*)d_in[6];              // (3H,)

    // scratch carve (all offsets naturally 256B-aligned)
    char* ws = (char*)d_ws;
    __bf16* xbf  = (__bf16*)ws;  ws += (size_t)T_DIM * N_DIM * I_DIM * 2;   // 32 MB
    __bf16* wihb = (__bf16*)ws;  ws += (size_t)3 * H_DIM * I_DIM * 2;      // 1.5 MB
    __bf16* whhb = (__bf16*)ws;  ws += (size_t)3 * H_DIM * H_DIM * 2;      // 1.5 MB
    float*  hf0  = (float*)ws;   ws += (size_t)N_DIM * H_DIM * 4;          // 512 KB
    float*  hf1  = (float*)ws;   ws += (size_t)N_DIM * H_DIM * 4;
    __bf16* hb0  = (__bf16*)ws;  ws += (size_t)N_DIM * H_DIM * 2;          // 256 KB
    __bf16* hb1  = (__bf16*)ws;  ws += (size_t)N_DIM * H_DIM * 2;
    __bf16* zrow = (__bf16*)ws;                                            // 1 KB zeros

    cvt_f32_to_bf16<<<4096, 256, 0, stream>>>(x,    xbf,  T_DIM * N_DIM * I_DIM);
    cvt_f32_to_bf16<<<1024, 256, 0, stream>>>(w_ih, wihb, 3 * H_DIM * I_DIM);
    cvt_f32_to_bf16<<<1024, 256, 0, stream>>>(w_hh, whhb, 3 * H_DIM * H_DIM);
    init_hidden<<<(N_DIM * H_DIM + 255) / 256, 256, 0, stream>>>(h0, hf0, hb0, zrow, N_DIM * H_DIM);

    float* out  = (float*)d_out;
    float* hfin = out + (size_t)T_DIM * N_DIM * H_DIM;

    float*  hf[2] = {hf0, hf1};
    __bf16* hb[2] = {hb0, hb1};

    // 512 tiles (16 row-tiles x 32 col-tiles), 1 wave each, 8 waves/block -> 64 blocks
    for (int t = 0; t < T_DIM; ++t) {
        gru_step_kernel<<<64, 256, 0, stream>>>(
            xbf, wihb, whhb, b_ih, b_hh, masks,
            hf[t & 1], hb[t & 1], zrow, hf[(t + 1) & 1], hb[(t + 1) & 1],
            out + (size_t)t * N_DIM * H_DIM,
            (t == T_DIM - 1) ? hfin : nullptr, t);
    }
}
